// Model_39676907883866
// MI455X (gfx1250) — compile-verified, matrix operand order checked by reading, not run on hardware
//
#include <hip/hip_runtime.h>
#include <hip/hip_bf16.h>

// ---------------------------------------------------------------------------
// Problem constants (from the reference)
// ---------------------------------------------------------------------------
#define BS    8
#define S     128
#define NHID  1024
#define NHEAD 8
#define DH    128      // NHID / NHEAD
#define L     1024     // S * NHEAD

typedef __bf16 bf16;
typedef __attribute__((ext_vector_type(16))) __bf16 v16bf;
typedef __attribute__((ext_vector_type(8)))  float  v8f;

// ---------------------------------------------------------------------------
// fp32 -> bf16 (round to nearest even), bit-level to avoid relying on native cvt
// ---------------------------------------------------------------------------
__device__ __forceinline__ bf16 f2bf(float x) {
    unsigned u = __float_as_uint(x);
    u += 0x7FFFu + ((u >> 16) & 1u);
    unsigned short h = (unsigned short)(u >> 16);
    return __builtin_bit_cast(bf16, h);
}

// ---------------------------------------------------------------------------
// WMMA bf16 fragment load.
// 16-bit A-matrix 16x32 layout (ISA 7.12.2): lane l holds row (l&15);
// vector elems 0..7  hold K = (l>>4)*8 + 0..7
// vector elems 8..15 hold K = 16 + (l>>4)*8 + 0..7
// Dense B (32x16) uses the same per-lane layout with N in place of M, so a
// B^T (N x K row-major) operand loads identically.
// ---------------------------------------------------------------------------
__device__ __forceinline__ v16bf load_frag(const bf16* tile, int ld) {
    const int lane = threadIdx.x & 31;
    const int r    = lane & 15;
    const int kb   = (lane >> 4) << 3;          // 0 or 8
    const bf16* p  = tile + (long)r * ld + kb;
    union { v16bf v; uint4 q[2]; } u;
    u.q[0] = *reinterpret_cast<const uint4*>(p);        // K = kb..kb+7
    u.q[1] = *reinterpret_cast<const uint4*>(p + 16);   // K = 16+kb..16+kb+7
    return u.v;
}

// ---------------------------------------------------------------------------
// Generic wave-tiled GEMM:  D(M x N) = A(M x K, row-major) * B^T(N x K, row-major)
// Block = 256 threads = 8 waves arranged WVM x WVN (WVM*WVN == 8).
// Wave tile = (16*TM) x (16*TN) WMMA tiles; TM+TN fragment loads feed TM*TN
// WMMAs per 32-wide K-step (register-level operand reuse).
// Map functor resolves per-z (batch/head) operand base pointers.
// Epi functor consumes each fp32 result element (fused bias / scale / store).
// ---------------------------------------------------------------------------
template <int TM, int TN, int WVM, int WVN, class Map, class Epi>
__global__ __launch_bounds__(256) void gemm_bt_kernel(Map map, int lda, int ldb,
                                                      int K, Epi epi) {
    const int z = blockIdx.z;
    const bf16* Ab;
    const bf16* Bb;
    map(z, Ab, Bb);

    const int wave = threadIdx.x >> 5;
    const int lane = threadIdx.x & 31;
    const int wm   = wave / WVN;
    const int wn   = wave % WVN;
    const int m0   = blockIdx.y * (WVM * TM * 16) + wm * (TM * 16);
    const int n0   = blockIdx.x * (WVN * TN * 16) + wn * (TN * 16);

    v8f acc[TM][TN];
    #pragma unroll
    for (int i = 0; i < TM; ++i)
        #pragma unroll
        for (int j = 0; j < TN; ++j) acc[i][j] = {};

    for (int k0 = 0; k0 < K; k0 += 32) {
        v16bf af[TM], bfv[TN];
        #pragma unroll
        for (int i = 0; i < TM; ++i)
            af[i] = load_frag(Ab + (long)(m0 + 16 * i) * lda + k0, lda);
        #pragma unroll
        for (int j = 0; j < TN; ++j)
            bfv[j] = load_frag(Bb + (long)(n0 + 16 * j) * ldb + k0, ldb);
        #pragma unroll
        for (int i = 0; i < TM; ++i)
            #pragma unroll
            for (int j = 0; j < TN; ++j)
                acc[i][j] = __builtin_amdgcn_wmma_f32_16x16x32_bf16(
                    false, af[i], false, bfv[j], (short)0, acc[i][j], false, false);
    }

    // C/D layout: lanes 0-15 -> M = vgpr, lanes 16-31 -> M = vgpr+8 ; N = lane&15
    const int rlo = (lane >> 4) << 3;
    const int c   = lane & 15;
    #pragma unroll
    for (int i = 0; i < TM; ++i)
        #pragma unroll
        for (int j = 0; j < TN; ++j)
            #pragma unroll
            for (int v = 0; v < 8; ++v)
                epi(z, m0 + 16 * i + v + rlo, n0 + 16 * j + c, acc[i][j][v]);
}

// ---------------------------------------------------------------------------
// Address-map functors
// ---------------------------------------------------------------------------
struct MapSplit {            // stage 1: A per batch (z>>3), B per head (z&7)
    const bf16* A; const bf16* B;
    __device__ void operator()(int z, const bf16*& a, const bf16*& b) const {
        a = A + (size_t)(z >> 3) * ((size_t)S * NHID);
        b = B + (size_t)(z & 7) * ((size_t)NHID * NHID);
    }
};
struct MapLin {              // A += z*sA ; B += z*sB
    const bf16* A; const bf16* B; long sA, sB;
    __device__ void operator()(int z, const bf16*& a, const bf16*& b) const {
        a = A + (size_t)z * sA;
        b = B + (size_t)z * sB;
    }
};

// ---------------------------------------------------------------------------
// Epilogue functors
// ---------------------------------------------------------------------------
struct EpiHeads {            // + b_h, write heads_r (z,L,DH) and headsT (z,DH,L)
    const float* bh; bf16* heads; bf16* headsT;
    __device__ void operator()(int z, int row, int col, float v) const {
        const int h = z & 7;
        v += bh[h * NHID + col];
        const int l = (row << 3) + (col >> 7);   // reshape (S,NHID)->(L,DH)
        const int d = col & 127;
        const bf16 x = f2bf(v);
        heads [((size_t)z * L + l) * DH + d] = x;
        headsT[((size_t)z * DH + d) * L + l] = x;
    }
};
struct EpiG {                // g in bf16
    bf16* g;
    __device__ void operator()(int z, int row, int col, float v) const {
        g[(size_t)z * L * L + (size_t)row * L + col] = f2bf(v);
    }
};
struct EpiW {                // w = v/sqrt(L) + x3[b], fp32 for softmax
    const float* x3; float* w;
    __device__ void operator()(int z, int row, int col, float v) const {
        const int b = z >> 3;
        w[(size_t)z * L * L + (size_t)row * L + col] =
            v * 0.03125f + x3[(size_t)b * L * L + (size_t)row * L + col];
    }
};
struct EpiX {                // out_h scattered into concat layout (b, L, NHID)
    bf16* xcat;
    __device__ void operator()(int z, int row, int col, float v) const {
        const int b = z >> 3, h = z & 7;
        xcat[((size_t)b * L + row) * NHID + h * DH + col] = f2bf(v);
    }
};
struct EpiOut {              // + b_o, fp32 final output
    const float* bo; float* out;
    __device__ void operator()(int z, int row, int col, float v) const {
        out[((size_t)z * L + row) * NHID + col] = v + bo[col];
    }
};

// ---------------------------------------------------------------------------
// fp32 -> bf16 convert
// ---------------------------------------------------------------------------
__global__ __launch_bounds__(256) void f32_to_bf16_kernel(
    const float* __restrict__ in, bf16* __restrict__ out, long n) {
    long i = (long)blockIdx.x * blockDim.x + threadIdx.x;
    if (i < n) out[i] = f2bf(in[i]);
}

// ---------------------------------------------------------------------------
// Row softmax: one 256-thread block per row of 1024, fp32 in -> bf16 out
// ---------------------------------------------------------------------------
__global__ __launch_bounds__(256) void softmax_kernel(
    const float* __restrict__ w, bf16* __restrict__ a) {
    const size_t row = blockIdx.x;
    const float* wr = w + row * L;
    bf16* ar = a + row * L;
    __shared__ float red[256];

    float vals[4];
    float m = -3.4e38f;
    #pragma unroll
    for (int j = 0; j < 4; ++j) {
        vals[j] = wr[threadIdx.x + 256 * j];
        m = fmaxf(m, vals[j]);
    }
    red[threadIdx.x] = m; __syncthreads();
    for (int s = 128; s > 0; s >>= 1) {
        if ((int)threadIdx.x < s) red[threadIdx.x] = fmaxf(red[threadIdx.x], red[threadIdx.x + s]);
        __syncthreads();
    }
    m = red[0]; __syncthreads();

    float sum = 0.f;
    #pragma unroll
    for (int j = 0; j < 4; ++j) { vals[j] = __expf(vals[j] - m); sum += vals[j]; }
    red[threadIdx.x] = sum; __syncthreads();
    for (int s = 128; s > 0; s >>= 1) {
        if ((int)threadIdx.x < s) red[threadIdx.x] += red[threadIdx.x + s];
        __syncthreads();
    }
    const float inv = 1.0f / red[0];
    #pragma unroll
    for (int j = 0; j < 4; ++j) ar[threadIdx.x + 256 * j] = f2bf(vals[j] * inv);
}

// ---------------------------------------------------------------------------
// Host driver
// ---------------------------------------------------------------------------
extern "C" void kernel_launch(void* const* d_in, const int* in_sizes, int n_in,
                              void* d_out, int out_size, void* d_ws, size_t ws_size,
                              hipStream_t stream) {
    // inputs: x1, x2, x3, W_h, b_h, W_o, b_o (all fp32)
    const float* x2 = (const float*)d_in[1];
    const float* x3 = (const float*)d_in[2];
    const float* Wh = (const float*)d_in[3];
    const float* bh = (const float*)d_in[4];
    const float* Wo = (const float*)d_in[5];
    const float* bo = (const float*)d_in[6];
    float* out = (float*)d_out;

    // workspace carve-up
    char* ws = (char*)d_ws;
    size_t off = 0;
    auto carve = [&](size_t bytes) {
        void* p = ws + off;
        off = (off + bytes + 255) & ~(size_t)255;
        return p;
    };
    const size_t E_X2  = (size_t)BS * S * NHID;           // 1,048,576
    const size_t E_WH  = (size_t)NHEAD * NHID * NHID;     // 8,388,608
    const size_t E_WO  = (size_t)NHID * NHID;             // 1,048,576
    const size_t E_HD  = (size_t)BS * NHEAD * L * DH;     // 8,388,608
    const size_t E_G   = (size_t)BS * NHEAD * L * L;      // 67,108,864

    bf16*  x2b    = (bf16*) carve(E_X2 * 2);
    bf16*  whb    = (bf16*) carve(E_WH * 2);
    bf16*  wob    = (bf16*) carve(E_WO * 2);
    bf16*  heads  = (bf16*) carve(E_HD * 2);
    bf16*  headsT = (bf16*) carve(E_HD * 2);
    bf16*  xcat   = (bf16*) carve((size_t)BS * L * NHID * 2);
    bf16*  gmat   = (bf16*) carve(E_G * 2);
    bf16*  amat   = (bf16*) carve(E_G * 2);
    float* wmat   = (float*)carve(E_G * 4);
    (void)ws_size; (void)in_sizes; (void)n_in; (void)out_size;

    // 0) fp32 -> bf16 operand conversion
    f32_to_bf16_kernel<<<(unsigned)((E_X2 + 255) / 256), 256, 0, stream>>>(x2, x2b, (long)E_X2);
    f32_to_bf16_kernel<<<(unsigned)((E_WH + 255) / 256), 256, 0, stream>>>(Wh, whb, (long)E_WH);
    f32_to_bf16_kernel<<<(unsigned)((E_WO + 255) / 256), 256, 0, stream>>>(Wo, wob, (long)E_WO);

    // 1) heads = x2[b] @ W_h[h]^T + b_h   (M=128, N=1024, K=1024) x 64
    //    wave tile 64x32, block 128x128
    gemm_bt_kernel<4, 2, 2, 4><<<dim3(8, 1, 64), 256, 0, stream>>>(
        MapSplit{x2b, whb}, NHID, NHID, NHID,
        EpiHeads{bh, heads, headsT});

    // 3) g = heads_r @ heads_r^T          (M=N=1024, K=128) x 64
    //    wave tile 64x64, block 128x256
    gemm_bt_kernel<4, 4, 2, 4><<<dim3(4, 8, 64), 256, 0, stream>>>(
        MapLin{heads, heads, (long)L * DH, (long)L * DH}, DH, DH, DH,
        EpiG{gmat});

    // 4) w = g @ g^T / 32 + x3            (M=N=K=1024) x 64   [dominant]
    //    wave tile 64x64, block 128x256
    gemm_bt_kernel<4, 4, 2, 4><<<dim3(4, 8, 64), 256, 0, stream>>>(
        MapLin{gmat, gmat, (long)L * L, (long)L * L}, L, L, L,
        EpiW{x3, wmat});

    // 5) a = softmax(w)
    softmax_kernel<<<BS * NHEAD * L, 256, 0, stream>>>(wmat, amat);

    // 6) out_h = a @ heads_r -> concat layout (M=1024, N=128, K=1024) x 64
    //    wave tile 64x64, block 256x128 (waves 4x2)
    gemm_bt_kernel<4, 4, 4, 2><<<dim3(1, 4, 64), 256, 0, stream>>>(
        MapLin{amat, headsT, (long)L * L, (long)DH * L}, L, L, L,
        EpiX{xcat});

    // 7) out = xcat @ W_o^T + b_o         (M=1024, N=1024, K=1024) x 8
    //    wave tile 64x64, block 128x256
    gemm_bt_kernel<4, 4, 2, 4><<<dim3(4, 8, BS), 256, 0, stream>>>(
        MapLin{xcat, wob, (long)L * NHID, 0L}, NHID, NHID, NHID,
        EpiOut{bo, out});
}
